// MultiLayerAttention_30365418783243
// MI455X (gfx1250) — matrix operand order repaired and checked by reading, not compiled, not run on hardware
//
#include <hip/hip_runtime.h>
#include <hip/hip_bf16.h>

// ---------------------------------------------------------------------------
// gfx1250 WMMA (wave32): D(16x16 f32) = A(16x32 bf16) x B(32x16 bf16) + C
// ---------------------------------------------------------------------------
typedef __attribute__((ext_vector_type(16))) __bf16 bf16x16;
typedef __attribute__((ext_vector_type(8)))  __bf16 bf16x8;
typedef __attribute__((ext_vector_type(8)))  float  f32x8;
typedef int v4i __attribute__((vector_size(16)));   // matches async builtin param

__device__ __forceinline__ bf16x16 pack16(const __bf16* p0, const __bf16* p1) {
    bf16x8 lo = *(const bf16x8*)p0;
    bf16x8 hi = *(const bf16x8*)p1;
    bf16x16 r;
#pragma unroll
    for (int i = 0; i < 8; ++i) { r[i] = lo[i]; r[i + 8] = hi[i]; }
    return r;
}

__device__ __forceinline__ f32x8 wmma_bf16(bf16x16 a, bf16x16 b, f32x8 c) {
    return __builtin_amdgcn_wmma_f32_16x16x32_bf16(false, a, false, b, (short)0, c,
                                                   false, false);
}

// 16B global -> LDS copy. Prefer the CDNA5 async-to-LDS path (ASYNCcnt) which
// bypasses the VGPR round trip; fall back to vector load/store.
#if defined(__HIP_DEVICE_COMPILE__) && __has_builtin(__builtin_amdgcn_global_load_async_to_lds_b128)
#define HAVE_ASYNC_LDS 1
#else
#define HAVE_ASYNC_LDS 0
#endif

__device__ __forceinline__ void copy16B_g2l(__bf16* ldst, const __bf16* gsrc) {
#if HAVE_ASYNC_LDS
    __builtin_amdgcn_global_load_async_to_lds_b128(
        (__attribute__((address_space(1))) v4i*)gsrc,
        (__attribute__((address_space(3))) v4i*)ldst, 0, 0);
#else
    *(bf16x8*)ldst = *(const bf16x8*)gsrc;
#endif
}

__device__ __forceinline__ void async_wait() {
#if HAVE_ASYNC_LDS
    asm volatile("s_wait_asynccnt 0" ::: "memory");
#endif
}

// ---------------------------------------------------------------------------
// f32 -> bf16 matrix conversion with stride repack (one-time pass; everything
// then lives bf16 in the 192MB L2).  grid: (ceil(cols/256), rows)
// ---------------------------------------------------------------------------
__global__ __launch_bounds__(256) void cvt_bf16_kernel(
    const float* __restrict__ src, __bf16* __restrict__ dst,
    int cols, int sld, int dld)
{
    const int c = blockIdx.x * 256 + threadIdx.x;
    const int r = blockIdx.y;
    if (c < cols) dst[(size_t)r * dld + c] = (__bf16)src[(size_t)r * sld + c];
}

// ---------------------------------------------------------------------------
// GEMM: C(MxN) = A(MxK,bf16) * B(bf16), f32 accumulate.
//   bTrans==0 : B is KxN row-major;  bTrans==1 : B is NxK row-major (A*B^T)
//   outBf16   : C element type (bf16 or f32)
// M multiple of 128; lda/ldb multiples of 8 (16B rows). N,K arbitrary.
// Block 256 thr (8 waves), tile 128x128x32, wave tile 32x64 (8 WMMA accums).
// ---------------------------------------------------------------------------
#define GLDT 40   // padded LDS row stride (elements): 32 + 8

__global__ __launch_bounds__(256) void gemm_bf16_kernel(
    const __bf16* __restrict__ A, int lda,
    const __bf16* __restrict__ B, int ldb,
    void* __restrict__ C, int ldc,
    int M, int N, int K, int bTrans, int outBf16)
{
    __shared__ __attribute__((aligned(16))) __bf16 As[128 * GLDT];
    __shared__ __attribute__((aligned(16))) __bf16 Bs[128 * GLDT];

    const int tid  = threadIdx.x;
    const int lane = tid & 31;
    const int wid  = tid >> 5;
    const int half = lane >> 4;
    const int l16  = lane & 15;

    const int m0 = blockIdx.y * 128;
    const int n0 = blockIdx.x * 128;
    const int wm = (wid >> 1) * 32;
    const int wn = (wid & 1) * 64;
    const bool nfull = (n0 + 128 <= N);

    f32x8 acc[2][4];
#pragma unroll
    for (int i = 0; i < 2; ++i)
#pragma unroll
        for (int j = 0; j < 4; ++j)
#pragma unroll
            for (int r = 0; r < 8; ++r) acc[i][j][r] = 0.0f;

    for (int k0 = 0; k0 < K; k0 += 32) {
        const bool kfull = (k0 + 32 <= K);

        // ---- stage A tile [row][k]: 128x32 bf16, 2 x 16B per thread ----
        if (kfull) {
#pragma unroll
            for (int p = 0; p < 2; ++p) {
                const int r  = (tid >> 2) + p * 64;
                const int kk = (tid & 3) * 8;
                copy16B_g2l(&As[r * GLDT + kk],
                            A + (size_t)(m0 + r) * lda + k0 + kk);
            }
        } else {
#pragma unroll
            for (int p = 0; p < 2; ++p) {
                const int r  = (tid >> 2) + p * 64;
                const int kk = (tid & 3) * 8;
                const size_t rb = (size_t)(m0 + r) * lda;
#pragma unroll
                for (int u = 0; u < 8; ++u) {
                    const int gk = k0 + kk + u;
                    __bf16 v = A[rb + (gk < K ? gk : K - 1)];   // clamped, branchless
                    if (gk >= K) v = (__bf16)0.0f;
                    As[r * GLDT + kk + u] = v;
                }
            }
        }

        // ---- stage B tile into [n][k] ----
        if (bTrans) {               // B is NxK row-major: k-contiguous rows
            if (kfull && nfull) {
#pragma unroll
                for (int p = 0; p < 2; ++p) {
                    const int n  = (tid >> 2) + p * 64;
                    const int kk = (tid & 3) * 8;
                    copy16B_g2l(&Bs[n * GLDT + kk],
                                B + (size_t)(n0 + n) * ldb + k0 + kk);
                }
            } else {
#pragma unroll
                for (int p = 0; p < 2; ++p) {
                    const int n  = (tid >> 2) + p * 64;
                    const int kk = (tid & 3) * 8;
                    const int gn = n0 + n;
                    const size_t rb = (size_t)(gn < N ? gn : N - 1) * ldb;
#pragma unroll
                    for (int u = 0; u < 8; ++u) {
                        const int gk = k0 + kk + u;
                        __bf16 v = B[rb + (gk < K ? gk : K - 1)];
                        if (gk >= K || gn >= N) v = (__bf16)0.0f;
                        Bs[n * GLDT + kk + u] = v;
                    }
                }
            }
        } else {                    // B is KxN row-major: transpose while staging
            if (kfull && nfull) {
#pragma unroll
                for (int p = 0; p < 2; ++p) {
                    const int kk = (tid >> 4) + p * 16;
                    const int n  = (tid & 15) * 8;
                    bf16x8 v = *(const bf16x8*)(B + (size_t)(k0 + kk) * ldb + n0 + n);
#pragma unroll
                    for (int u = 0; u < 8; ++u) Bs[(n + u) * GLDT + kk] = v[u];
                }
            } else {
#pragma unroll
                for (int p = 0; p < 2; ++p) {
                    const int kk = (tid >> 4) + p * 16;
                    const int n  = (tid & 15) * 8;
                    const int gk = k0 + kk;
                    const size_t rb = (size_t)(gk < K ? gk : K - 1) * ldb;
#pragma unroll
                    for (int u = 0; u < 8; ++u) {
                        const int gn = n0 + n + u;
                        __bf16 v = B[rb + (gn < N ? gn : N - 1)];
                        if (gk >= K || gn >= N) v = (__bf16)0.0f;
                        Bs[(n + u) * GLDT + kk] = v;
                    }
                }
            }
        }
        async_wait();
        __syncthreads();

        // ---- 2 A-frags, 4 B-frags, 8 WMMAs ----
        bf16x16 af[2], bfm[4];
#pragma unroll
        for (int i = 0; i < 2; ++i) {
            const __bf16* ap = &As[(wm + 16 * i + l16) * GLDT + half * 8];
            af[i] = pack16(ap, ap + 16);
        }
#pragma unroll
        for (int j = 0; j < 4; ++j) {
            const __bf16* bp = &Bs[(wn + 16 * j + l16) * GLDT + half * 8];
            bfm[j] = pack16(bp, bp + 16);
        }
#pragma unroll
        for (int i = 0; i < 2; ++i)
#pragma unroll
            for (int j = 0; j < 4; ++j)
                acc[i][j] = wmma_bf16(af[i], bfm[j], acc[i][j]);
        __syncthreads();
    }

    // ---- epilogue ----
#pragma unroll
    for (int i = 0; i < 2; ++i)
#pragma unroll
        for (int j = 0; j < 4; ++j) {
            const int nn = n0 + wn + 16 * j + l16;
            if (nn < N) {
                const int mbase = m0 + wm + 16 * i + half * 8;
                if (outBf16) {
                    __bf16* Cb = (__bf16*)C;
#pragma unroll
                    for (int r = 0; r < 8; ++r)
                        Cb[(size_t)(mbase + r) * ldc + nn] = (__bf16)acc[i][j][r];
                } else {
                    float* Cf = (float*)C;
#pragma unroll
                    for (int r = 0; r < 8; ++r)
                        Cf[(size_t)(mbase + r) * ldc + nn] = acc[i][j][r];
                }
            }
        }
}

// ---------------------------------------------------------------------------
// Row LayerNorm (f32 stats), bf16 output. One block per row.
// ---------------------------------------------------------------------------
__global__ __launch_bounds__(256) void ln_kernel(
    const float* __restrict__ src, __bf16* __restrict__ dst,
    const float* __restrict__ w, const float* __restrict__ b,
    int C, int srcStride, int dstStride)
{
    const int row  = blockIdx.x;
    const int lane = threadIdx.x & 31;
    const int wid  = threadIdx.x >> 5;
    const float* x = src + (size_t)row * srcStride;
    __bf16* y      = dst + (size_t)row * dstStride;

    float s = 0.f, s2 = 0.f;
    for (int i = threadIdx.x; i < C; i += 256) {
        float v = x[i]; s += v; s2 += v * v;
    }
#pragma unroll
    for (int mk = 16; mk >= 1; mk >>= 1) {
        s  += __shfl_xor(s,  mk, 32);
        s2 += __shfl_xor(s2, mk, 32);
    }
    __shared__ float rs[8], rs2[8];
    if (lane == 0) { rs[wid] = s; rs2[wid] = s2; }
    __syncthreads();
    float ts = 0.f, ts2 = 0.f;
#pragma unroll
    for (int i = 0; i < 8; ++i) { ts += rs[i]; ts2 += rs2[i]; }
    const float mean = ts / C;
    const float var  = ts2 / C - mean * mean;
    const float inv  = rsqrtf(var + 1e-5f);
    for (int i = threadIdx.x; i < C; i += 256)
        y[i] = (__bf16)((x[i] - mean) * inv * w[i] + b[i]);
}

// ---------------------------------------------------------------------------
// RoPE: Q (f32, in place on dims [h*128+64, h*128+128)); Kr -> bf16 buffer.
// ---------------------------------------------------------------------------
__global__ __launch_bounds__(256) void rope_kernel(
    float* __restrict__ Q, const float* __restrict__ ckv, __bf16* __restrict__ krope)
{
    const int idx = blockIdx.x * 256 + threadIdx.x;
    const int i   = idx & 31;            // pair index 0..31
    const int hh  = (idx >> 5) % 17;     // 0..15 heads, 16 => Kr
    const int ri  = idx / (32 * 17);     // token row 0..4095
    if (ri >= 4096) return;
    const int pos = ri & 2047;
    const float freq = powf(10000.0f, -(2.0f * i) / 128.0f);
    const float ang  = (float)pos * freq;
    const float c = cosf(ang), sn = sinf(ang);
    if (hh < 16) {
        float* base = Q + (size_t)ri * 2048 + hh * 128 + 64;
        const float x1 = base[i], x2 = base[i + 32];
        base[i]      = x1 * c - x2 * sn;
        base[i + 32] = x2 * c + x1 * sn;
    } else {
        const float* kb = ckv + (size_t)ri * 1429 + 1365;
        __bf16* o = krope + (size_t)ri * 64;
        const float x1 = kb[i], x2 = kb[i + 32];
        o[i]      = (__bf16)(x1 * c - x2 * sn);
        o[i + 32] = (__bf16)(x2 * c + x1 * sn);
    }
}

// ---------------------------------------------------------------------------
// Causal flash attention. 128 queries/block, 16/wave. bf16 K/V streams.
//   Q  : f32 [4096][2048] (roped)     KV : bf16 [4096][3072]
//   KR : bf16 [4096][64]              O  : bf16 [4096][2048]
// ---------------------------------------------------------------------------
#define KLD 136   // Ks stride: 128 + 8
#define VLD 40    // Vs stride: 32 + 8
#define PLD 40    // Ps stride: 32 + 8

__global__ __launch_bounds__(256) void attn_kernel(
    const float* __restrict__ Q, const __bf16* __restrict__ KV,
    const __bf16* __restrict__ KR, __bf16* __restrict__ O)
{
    __shared__ __attribute__((aligned(16))) __bf16 Ks[32 * KLD];   // [key][dim]
    __shared__ __attribute__((aligned(16))) __bf16 Vs[128 * VLD];  // [vdim][key]
    __shared__ __attribute__((aligned(16))) __bf16 Ps[8 * 16 * PLD];

    const int tid  = threadIdx.x;
    const int lane = tid & 31;
    const int wid  = tid >> 5;
    const int half = lane >> 4;
    const int l16  = lane & 15;

    const int bh = blockIdx.y;
    const int b  = bh >> 4, h = bh & 15;
    const int qblk = blockIdx.x;
    const int q0   = qblk * 128 + wid * 16;
    const size_t rowB = (size_t)b * 2048;

    // preload Q fragments (4 A-frags over head dim 0..127), f32 -> bf16 once
    bf16x16 qf[4];
    {
        const float* qp = Q + (rowB + q0 + l16) * 2048 + h * 128 + half * 8;
#pragma unroll
        for (int kk = 0; kk < 4; ++kk) {
            const float* p0 = qp + kk * 32;
            float buf[16];
            *(float4*)(buf + 0)  = *(const float4*)(p0);
            *(float4*)(buf + 4)  = *(const float4*)(p0 + 4);
            *(float4*)(buf + 8)  = *(const float4*)(p0 + 16);
            *(float4*)(buf + 12) = *(const float4*)(p0 + 20);
            bf16x16 f;
#pragma unroll
            for (int u = 0; u < 16; ++u) f[u] = (__bf16)buf[u];
            qf[kk] = f;
        }
    }

    f32x8 acc[8];
#pragma unroll
    for (int j = 0; j < 8; ++j)
#pragma unroll
        for (int r = 0; r < 8; ++r) acc[j][r] = 0.0f;
    float mrow[8], lrow[8];
#pragma unroll
    for (int r = 0; r < 8; ++r) { mrow[r] = -1e30f; lrow[r] = 0.0f; }

    const float scale = 0.08838834764831845f;   // 1/sqrt(128)
    const int nkb = (qblk + 1) * 4;

    for (int kb = 0; kb < nkb; ++kb) {
        const int t0 = kb * 32;
        // ---- stage K tile [key][dim] via 16B copies (async-to-LDS path) ----
        {
            const int chunk = tid & 15;          // 16 chunks of 8 dims
            const int key0  = tid >> 4;          // 16 keys per pass
#pragma unroll
            for (int p = 0; p < 2; ++p) {
                const int ky = key0 + p * 16;
                const int t  = t0 + ky;
                const int d  = chunk * 8;
                const __bf16* src = (d < 64)
                    ? KV + (rowB + t) * 3072 + h * 192 + d
                    : KR + (rowB + t) * 64 + (d - 64);
                copy16B_g2l(&Ks[ky * KLD + d], src);
            }
        }
        // ---- stage V tile transposed [vdim][key] ----
        {
            const int chunk = tid & 15;
            const int key0  = tid >> 4;
#pragma unroll
            for (int p = 0; p < 2; ++p) {
                const int ky = key0 + p * 16;
                const int t  = t0 + ky;
                const int d  = chunk * 8;
                bf16x8 v = *(const bf16x8*)(KV + (rowB + t) * 3072 + h * 192 + 64 + d);
#pragma unroll
                for (int u = 0; u < 8; ++u) Vs[(d + u) * VLD + ky] = v[u];
            }
        }
        async_wait();
        __syncthreads();

        // ---- QK^T: 16x32 scores, K-dim = 128 head dims ----
        f32x8 sc[2];
#pragma unroll
        for (int j = 0; j < 2; ++j) {
#pragma unroll
            for (int r = 0; r < 8; ++r) sc[j][r] = 0.0f;
#pragma unroll
            for (int kk = 0; kk < 4; ++kk) {
                const __bf16* kp = &Ks[(16 * j + l16) * KLD + kk * 32 + half * 8];
                sc[j] = wmma_bf16(qf[kk], pack16(kp, kp + 16), sc[j]);
            }
        }
        // ---- scale + causal mask ----
#pragma unroll
        for (int j = 0; j < 2; ++j) {
            const int t = t0 + 16 * j + l16;
#pragma unroll
            for (int r = 0; r < 8; ++r) {
                const int q = q0 + r + 8 * half;
                const float s = sc[j][r] * scale;
                sc[j][r] = (t <= q) ? s : -1e30f;
            }
        }
        // ---- online softmax ----
        float rmax[8];
#pragma unroll
        for (int r = 0; r < 8; ++r) rmax[r] = fmaxf(sc[0][r], sc[1][r]);
#pragma unroll
        for (int mk = 8; mk >= 1; mk >>= 1)
#pragma unroll
            for (int r = 0; r < 8; ++r)
                rmax[r] = fmaxf(rmax[r], __shfl_xor(rmax[r], mk, 32));
        float alpha[8];
#pragma unroll
        for (int r = 0; r < 8; ++r) {
            const float mn = fmaxf(mrow[r], rmax[r]);
            alpha[r] = __expf(mrow[r] - mn);
            mrow[r]  = mn;
        }
        float rsum[8];
#pragma unroll
        for (int r = 0; r < 8; ++r) rsum[r] = 0.0f;
        __bf16* pw = &Ps[wid * 16 * PLD];
#pragma unroll
        for (int j = 0; j < 2; ++j)
#pragma unroll
            for (int r = 0; r < 8; ++r) {
                const float p = __expf(sc[j][r] - mrow[r]);
                rsum[r] += p;
                pw[(r + 8 * half) * PLD + 16 * j + l16] = (__bf16)p;
            }
#pragma unroll
        for (int mk = 8; mk >= 1; mk >>= 1)
#pragma unroll
            for (int r = 0; r < 8; ++r) rsum[r] += __shfl_xor(rsum[r], mk, 32);
#pragma unroll
        for (int r = 0; r < 8; ++r) lrow[r] = lrow[r] * alpha[r] + rsum[r];
#pragma unroll
        for (int j = 0; j < 8; ++j)
#pragma unroll
            for (int r = 0; r < 8; ++r) acc[j][r] *= alpha[r];

        // same-wave LDS store->load ordering for P tile (CDNA5 split counter)
        asm volatile("s_wait_dscnt 0" ::: "memory");

        // ---- PV: out(16x128) += P(16x32) x V(32x128) ----
        const __bf16* pp = &pw[l16 * PLD + half * 8];
        bf16x16 pf = pack16(pp, pp + 16);
#pragma unroll
        for (int j = 0; j < 8; ++j) {
            const __bf16* vp = &Vs[(16 * j + l16) * VLD + half * 8];
            acc[j] = wmma_bf16(pf, pack16(vp, vp + 16), acc[j]);
        }
        __syncthreads();
    }

    // ---- epilogue: normalize, write bf16 ----
#pragma unroll
    for (int j = 0; j < 8; ++j) {
        const int n = h * 128 + 16 * j + l16;
#pragma unroll
        for (int r = 0; r < 8; ++r) {
            const int q = q0 + r + 8 * half;
            O[(rowB + q) * 2048 + n] = (__bf16)(acc[j][r] / lrow[r]);
        }
    }
}

// ---------------------------------------------------------------------------
// Launcher
// ---------------------------------------------------------------------------
extern "C" void kernel_launch(void* const* d_in, const int* in_sizes, int n_in,
                              void* d_out, int out_size, void* d_ws, size_t ws_size,
                              hipStream_t stream) {
    (void)in_sizes; (void)n_in; (void)out_size; (void)ws_size;
    const float* query   = (const float*)d_in[0];
    const float* W_dq    = (const float*)d_in[1];
    const float* W_uq    = (const float*)d_in[2];
    const float* q_ln_w  = (const float*)d_in[3];
    const float* q_ln_b  = (const float*)d_in[4];
    const float* W_dkv   = (const float*)d_in[5];
    const float* W_ukv   = (const float*)d_in[6];
    const float* kv_ln_w = (const float*)d_in[7];
    const float* kv_ln_b = (const float*)d_in[8];
    const float* W_o     = (const float*)d_in[9];

    float* out = (float*)d_out;                    // [4096][2048] f32
    float* ckv = out + (size_t)4096 * 2048;        // [4096][1429] f32 (2nd output)

    char* w = (char*)d_ws;
    auto alloc = [&](size_t bytes) -> char* {
        char* p = w; w += (bytes + 255) & ~(size_t)255; return p;
    };
    __bf16* queryb = (__bf16*)alloc((size_t)4096 * 2048 * 2);
    __bf16* Wdqb   = (__bf16*)alloc((size_t)2048 * 1024 * 2);
    __bf16* Wuqb   = (__bf16*)alloc((size_t)1024 * 2048 * 2);
    __bf16* Wdkvb  = (__bf16*)alloc((size_t)2048 * 1432 * 2);  // padded 1429->1432
    __bf16* Wukvb  = (__bf16*)alloc((size_t)1365 * 3072 * 2);
    __bf16* Wob    = (__bf16*)alloc((size_t)2048 * 2048 * 2);
    float*  cqf    = (float*) alloc((size_t)4096 * 1024 * 4);
    __bf16* cqb    = (__bf16*)alloc((size_t)4096 * 1024 * 2);
    __bf16* kvlatb = (__bf16*)alloc((size_t)4096 * 1376 * 2);  // padded 1365->1376
    float*  Qf     = (float*) alloc((size_t)4096 * 2048 * 4);
    __bf16* KVbb   = (__bf16*)alloc((size_t)4096 * 3072 * 2);
    __bf16* KRb    = (__bf16*)alloc((size_t)4096 * 64 * 2);
    __bf16* attnob = (__bf16*)alloc((size_t)4096 * 2048 * 2);

    dim3 blk(256);

    // 0) one-time f32 -> bf16 conversions (all streams then bf16, L2-resident)
    cvt_bf16_kernel<<<dim3(8, 4096),  blk, 0, stream>>>(query, queryb, 2048, 2048, 2048);
    cvt_bf16_kernel<<<dim3(4, 2048),  blk, 0, stream>>>(W_dq,  Wdqb,   1024, 1024, 1024);
    cvt_bf16_kernel<<<dim3(8, 1024),  blk, 0, stream>>>(W_uq,  Wuqb,   2048, 2048, 2048);
    cvt_bf16_kernel<<<dim3(6, 2048),  blk, 0, stream>>>(W_dkv, Wdkvb,  1429, 1429, 1432);
    cvt_bf16_kernel<<<dim3(12, 1365), blk, 0, stream>>>(W_ukv, Wukvb,  3072, 3072, 3072);
    cvt_bf16_kernel<<<dim3(8, 2048),  blk, 0, stream>>>(W_o,   Wob,    2048, 2048, 2048);

    // 1) down-projections
    gemm_bf16_kernel<<<dim3(8, 32),  blk, 0, stream>>>(queryb, 2048, Wdqb,  1024, cqf, 1024, 4096, 1024, 2048, 0, 0);
    gemm_bf16_kernel<<<dim3(12, 32), blk, 0, stream>>>(queryb, 2048, Wdkvb, 1432, ckv, 1429, 4096, 1429, 2048, 0, 0);

    // 2) layernorms -> bf16 activations
    ln_kernel<<<4096, blk, 0, stream>>>(cqf, cqb,    q_ln_w,  q_ln_b,  1024, 1024, 1024);
    ln_kernel<<<4096, blk, 0, stream>>>(ckv, kvlatb, kv_ln_w, kv_ln_b, 1365, 1429, 1376);

    // 3) up-projections (Q in f32 for RoPE; KV straight to bf16)
    gemm_bf16_kernel<<<dim3(16, 32), blk, 0, stream>>>(cqb,    1024, Wuqb,  2048, Qf,   2048, 4096, 2048, 1024, 0, 0);
    gemm_bf16_kernel<<<dim3(24, 32), blk, 0, stream>>>(kvlatb, 1376, Wukvb, 3072, KVbb, 3072, 4096, 3072, 1365, 0, 1);

    // 4) RoPE on Q (in place) and Kr -> bf16
    rope_kernel<<<8704, blk, 0, stream>>>(Qf, ckv, KRb);

    // 5) causal flash attention -> bf16 [4096][2048]
    attn_kernel<<<dim3(16, 32), blk, 0, stream>>>(Qf, KVbb, KRb, attnob);

    // 6) output projection: out = attn_out @ W_o^T (f32 result)
    gemm_bf16_kernel<<<dim3(16, 32), blk, 0, stream>>>(attnob, 2048, Wob, 2048, out, 2048, 4096, 2048, 2048, 1, 0);
}
